// ImprovedCrossModalAttention_2929167696341
// MI455X (gfx1250) — matrix operand order repaired and checked by reading, not compiled, not run on hardware
//
#include <hip/hip_runtime.h>
#include <hip/hip_bf16.h>
#include <math.h>

// ---------------------------------------------------------------------------
// Sizes (compile-time, from the reference)
// ---------------------------------------------------------------------------
#define DMODEL 2048
#define NHEAD  16
#define HD     128          // DMODEL / NHEAD
#define MMOD   8
#define SLEN   8
#define NEXP   8
#define TOPK   3

typedef __attribute__((ext_vector_type(16))) _Float16 v16h;
typedef __attribute__((ext_vector_type(8)))  float    v8f;
typedef __attribute__((ext_vector_type(4)))  float    f32x4;

// ---------------------------------------------------------------------------
// Workspace layout (float offsets into d_ws).  comb/h1/h2 are padded to 16
// rows so the WMMA A-fragment loads are unconditional (padding rows only
// pollute padding rows of the outputs, which are never consumed).
// h1/h2/qkv are contiguous so one zero-fill covers all split-K accumulators.
// ---------------------------------------------------------------------------
#define OFF_SUMM   0                         // 8*2048
#define OFF_COMB   16384                     // 16*4096 (rows 8..15 = padding)
#define OFF_H1     81920                     // 16*2048 (split-K accumulator)
#define OFF_H2     114688                    // 16*1024 (split-K accumulator)
#define OFF_QKV    131072                    // 64*6144 (split-K accumulator)
#define OFF_MASK   524288                    // 64
#define OFF_X      524352                    // 64*2048
#define OFF_APRE   655424                    // 2048
#define OFF_ATT    657472                    // 2048
#define OFF_G1     659520                    // 1024
#define OFF_TOPI   660544                    // 3 ints
#define OFF_GATES  660548                    // 3 (+1 pad)
#define OFF_HH     660552                    // 3*4096 (16B aligned)
#define OFF_Y      672840                    // 3*2048
#define ZERO_BASE  OFF_H1
#define ZERO_LEN   (32768 + 16384 + 393216)  // h1+h2+qkv = 442368 floats

// ---------------------------------------------------------------------------
// Zero-fill (float4 stores) for the split-K accumulators, every launch.
// ---------------------------------------------------------------------------
__global__ __launch_bounds__(256)
void zero_kernel(float* __restrict__ p, int n4)
{
    int i = blockIdx.x * 256 + threadIdx.x;
    if (i < n4) {
        f32x4 z = {0.f, 0.f, 0.f, 0.f};
        reinterpret_cast<f32x4*>(p)[i] = z;
    }
}

// ---------------------------------------------------------------------------
// Kernel 0: x = feats + mod_emb ; summ = mean_s x ; comb = [summ | context]
// ---------------------------------------------------------------------------
__global__ __launch_bounds__(256)
void prep_kernel(const float* __restrict__ feats, const float* __restrict__ context,
                 const float* __restrict__ mod_emb,
                 float* __restrict__ x, float* __restrict__ summ, float* __restrict__ comb)
{
    int idx = blockIdx.x * 256 + threadIdx.x;        // 0 .. M*D-1
    int m = idx >> 11;
    int d = idx & (DMODEL - 1);
    float me = mod_emb[idx];
    float acc = 0.f;
    #pragma unroll
    for (int s = 0; s < SLEN; ++s) {
        float v = feats[((size_t)m * SLEN + s) * DMODEL + d] + me;
        x[((size_t)m * SLEN + s) * DMODEL + d] = v;
        acc += v;
    }
    float mean = acc * (1.0f / SLEN);
    summ[idx] = mean;
    comb[(size_t)m * (2 * DMODEL) + d] = mean;
    comb[(size_t)m * (2 * DMODEL) + DMODEL + d] = context[d];
}

// ---------------------------------------------------------------------------
// Split-K WMMA token-GEMM partial:  C[MT*16,N] += A[MT*16,kchunk] * W^T slice.
// grid = (N/16, KSPLIT); one wave per (16-col tile, K-slice).  Branchless
// inner loop: 8x global_load_b128 (weights non-temporal) -> cvt_pk_f16 ->
// v_wmma_f32_16x16x32_f16, MT row-tiles reuse each streamed B fragment.
// fp32 atomicAdd accumulation; bias+activation applied by a later epilogue.
// ---------------------------------------------------------------------------
template<int MT>
__global__ __launch_bounds__(32)
void wmma_tokgemm(const float* __restrict__ A, int Kdim, int kChunk,
                  const float* __restrict__ W, float* __restrict__ C, int N)
{
    const int lane = threadIdx.x;
    const int lo   = lane & 15;
    const bool hi  = lane >= 16;
    const int n    = blockIdx.x * 16 + lo;
    const int kStart = blockIdx.y * kChunk;
    const int kEnd   = kStart + kChunk;
    const float* __restrict__ wrow = W + (size_t)n * Kdim + (hi ? 16 : 0);
    const int aoff = hi ? 16 : 0;

    v8f acc[MT];
    #pragma unroll
    for (int m = 0; m < MT; ++m)
        #pragma unroll
        for (int v = 0; v < 8; ++v) acc[m][v] = 0.0f;

    for (int k0 = kStart; k0 < kEnd; k0 += 32) {
        // ---- B fragment: 16 contiguous floats of weight row n (streaming/NT)
        const f32x4* wp = reinterpret_cast<const f32x4*>(wrow + k0);
        f32x4 b0 = __builtin_nontemporal_load(wp + 0);
        f32x4 b1 = __builtin_nontemporal_load(wp + 1);
        f32x4 b2 = __builtin_nontemporal_load(wp + 2);
        f32x4 b3 = __builtin_nontemporal_load(wp + 3);
        v16h bf;
        bf[0]=(_Float16)b0[0]; bf[1]=(_Float16)b0[1]; bf[2]=(_Float16)b0[2]; bf[3]=(_Float16)b0[3];
        bf[4]=(_Float16)b1[0]; bf[5]=(_Float16)b1[1]; bf[6]=(_Float16)b1[2]; bf[7]=(_Float16)b1[3];
        bf[8]=(_Float16)b2[0]; bf[9]=(_Float16)b2[1]; bf[10]=(_Float16)b2[2]; bf[11]=(_Float16)b2[3];
        bf[12]=(_Float16)b3[0]; bf[13]=(_Float16)b3[1]; bf[14]=(_Float16)b3[2]; bf[15]=(_Float16)b3[3];

        #pragma unroll
        for (int m = 0; m < MT; ++m) {
            // A fragment (activations, L2-resident, always in-bounds: padded rows)
            const float* ap = A + (size_t)(m * 16 + lo) * Kdim + k0 + aoff;
            const f32x4* a01 = reinterpret_cast<const f32x4*>(ap);
            const f32x4* a23 = reinterpret_cast<const f32x4*>(ap + 8);
            f32x4 a0 = a01[0], a1 = a01[1];      // K rel aoff+0..7   -> halves 0..7
            f32x4 a2 = a23[0], a3 = a23[1];      // K rel aoff+8..15  -> halves 8..15
            v16h af;
            af[0]=(_Float16)a0[0]; af[1]=(_Float16)a0[1]; af[2]=(_Float16)a0[2]; af[3]=(_Float16)a0[3];
            af[4]=(_Float16)a1[0]; af[5]=(_Float16)a1[1]; af[6]=(_Float16)a1[2]; af[7]=(_Float16)a1[3];
            af[8]=(_Float16)a2[0]; af[9]=(_Float16)a2[1]; af[10]=(_Float16)a2[2]; af[11]=(_Float16)a2[3];
            af[12]=(_Float16)a3[0]; af[13]=(_Float16)a3[1]; af[14]=(_Float16)a3[2]; af[15]=(_Float16)a3[3];

            acc[m] = __builtin_amdgcn_wmma_f32_16x16x32_f16(
                false, af, false, bf, (short)0, acc[m], false, false);
        }
    }

    #pragma unroll
    for (int m = 0; m < MT; ++m)
        #pragma unroll
        for (int v = 0; v < 8; ++v) {
            int row = m * 16 + v + (hi ? 8 : 0);
            atomicAdd(&C[(size_t)row * N + n], acc[m][v]);
        }
}

// ---------------------------------------------------------------------------
// Epilogue after split-K: C[i] = act(C[i] + bias[i % N])
// ---------------------------------------------------------------------------
__global__ __launch_bounds__(256)
void bias_act_kernel(float* __restrict__ C, const float* __restrict__ bias,
                     int N, int total, int act)
{
    int i = blockIdx.x * 256 + threadIdx.x;
    if (i >= total) return;
    int n = i % N;
    float v = C[i] + bias[n];
    if (act == 1) v = fmaxf(v, 0.0f);
    C[i] = v;
}

// ---------------------------------------------------------------------------
// Routing tail: rw = normalize(sigmoid(h2 @ rW3^T + rb3)); quantile mask.
// Also zero-inits the attended_pre accumulator. One block, 64 threads.
// ---------------------------------------------------------------------------
__global__ __launch_bounds__(64)
void routing_finish(const float* __restrict__ h2, const float* __restrict__ rW3,
                    const float* __restrict__ rb3, float* __restrict__ mask,
                    float* __restrict__ rw_out, float* __restrict__ attended_pre)
{
    __shared__ float part[64];
    __shared__ float rw[MMOD];
    __shared__ float wv[64];
    __shared__ float thr_s;
    int t = threadIdx.x;
    int m = t >> 3, c = t & 7;
    float s = 0.f;
    for (int k = c * 128; k < c * 128 + 128; ++k)
        s += h2[m * 1024 + k] * rW3[k];
    part[t] = s;
    __syncthreads();
    if (c == 0) {
        float d = 0.f;
        for (int j = 0; j < 8; ++j) d += part[m * 8 + j];
        d += rb3[0];
        rw[m] = 1.0f / (1.0f + expf(-d));
    }
    __syncthreads();
    if (t == 0) {
        float tot = 0.f;
        for (int i = 0; i < MMOD; ++i) tot += rw[i];
        for (int i = 0; i < MMOD; ++i) rw[i] /= tot;
    }
    __syncthreads();
    if (t < MMOD) rw_out[t] = rw[t];
    wv[t] = rw[t >> 3] * rw[t & 7];
    __syncthreads();
    if (t == 0) {
        float a[64];
        for (int i = 0; i < 64; ++i) a[i] = wv[i];
        for (int i = 1; i < 64; ++i) {               // insertion sort (64 vals)
            float key = a[i]; int j = i - 1;
            while (j >= 0 && a[j] > key) { a[j + 1] = a[j]; --j; }
            a[j + 1] = key;
        }
        // quantile(0.9): pos = 0.9*63 = 56.7 -> 0.3*a[56] + 0.7*a[57]
        thr_s = a[56] * 0.3f + a[57] * 0.7f;
    }
    __syncthreads();
    mask[t] = (wv[t] < thr_s) ? -1e9f : 0.0f;
    for (int i = t; i < DMODEL; i += 64) attended_pre[i] = 0.0f;
}

// ---------------------------------------------------------------------------
// Attention for q at s=0 only; accumulates mean_m (attn @ v) into attended_pre.
// grid = M*H blocks, 128 threads (= head dim).
// ---------------------------------------------------------------------------
__global__ __launch_bounds__(128)
void attention_kernel(const float* __restrict__ qkv, const float* __restrict__ mask,
                      float* __restrict__ attended_pre)
{
    __shared__ float red[128];
    __shared__ float sc[SLEN];
    const int m = blockIdx.x >> 4;
    const int h = blockIdx.x & 15;
    const int t = threadIdx.x;
    const int ld = 3 * DMODEL;
    const float qv = qkv[(size_t)(m * SLEN + 0) * ld + h * HD + t];
    for (int k = 0; k < SLEN; ++k) {
        float p = qv * qkv[(size_t)(m * SLEN + k) * ld + DMODEL + h * HD + t];
        red[t] = p; __syncthreads();
        for (int off = 64; off > 0; off >>= 1) {
            if (t < off) red[t] += red[t + off];
            __syncthreads();
        }
        if (t == 0) sc[k] = red[0] * 0.08838834764831845f + mask[k];  // 1/sqrt(128)
        __syncthreads();
    }
    if (t == 0) {
        float mx = sc[0];
        for (int k = 1; k < SLEN; ++k) mx = fmaxf(mx, sc[k]);
        float sum = 0.f;
        for (int k = 0; k < SLEN; ++k) { sc[k] = expf(sc[k] - mx); sum += sc[k]; }
        float inv = 1.0f / sum;
        for (int k = 0; k < SLEN; ++k) sc[k] *= inv;
    }
    __syncthreads();
    float o = 0.f;
    for (int k = 0; k < SLEN; ++k)
        o += sc[k] * qkv[(size_t)(m * SLEN + k) * ld + 2 * DMODEL + h * HD + t];
    atomicAdd(&attended_pre[h * HD + t], o * (1.0f / MMOD));
}

// ---------------------------------------------------------------------------
// Generic fp32 GEMV: y[b][n] = act( x[b] . Wrow(n of expert e) + bias )
// 8 waves/block, one wave per output row, NT float4 weight streaming +
// wave32 shuffle reduction.  expIdx==nullptr -> e=0. act: 0/1 relu/2 gelu.
// ---------------------------------------------------------------------------
__global__ __launch_bounds__(256)
void gemv_kernel(const float* __restrict__ x, long long xStride,
                 const float* __restrict__ Wb, long long wEStride,
                 const float* __restrict__ bb, long long bEStride,
                 const int* __restrict__ expIdx,
                 float* __restrict__ y, long long yStride,
                 int N, int Kd, int act)
{
    const int wave = threadIdx.x >> 5;
    const int lane = threadIdx.x & 31;
    const int n = blockIdx.x * 8 + wave;
    const int b = blockIdx.y;
    if (n >= N) return;
    const long long e = expIdx ? (long long)expIdx[b] : 0;
    const float* __restrict__ wrow = Wb + e * wEStride + (long long)n * Kd;
    const float* __restrict__ xv   = x + (long long)b * xStride;
    float s = 0.f;
    for (int k = lane * 4; k < Kd; k += 128) {
        f32x4 w4 = __builtin_nontemporal_load(reinterpret_cast<const f32x4*>(wrow + k));
        f32x4 x4 = *reinterpret_cast<const f32x4*>(xv + k);
        s += w4[0] * x4[0] + w4[1] * x4[1] + w4[2] * x4[2] + w4[3] * x4[3];
    }
    for (int off = 16; off > 0; off >>= 1) s += __shfl_down(s, off, 32);
    if (lane == 0) {
        float bias = bb ? (bb + e * bEStride)[n] : 0.0f;
        float v = s + bias;
        if (act == 1)      v = fmaxf(v, 0.0f);
        else if (act == 2) v = 0.5f * v * (1.0f + erff(v * 0.70710678118654752f));
        y[(long long)b * yStride + n] = v;
    }
}

// ---------------------------------------------------------------------------
// Gate: logits = g1 @ gW2^T + gb2 ; softmax ; top-3 ; renormalized gates.
// ---------------------------------------------------------------------------
__global__ __launch_bounds__(64)
void gate_topk(const float* __restrict__ g1, const float* __restrict__ gW2,
               const float* __restrict__ gb2,
               int* __restrict__ top_i, float* __restrict__ gates)
{
    __shared__ float part[64];
    __shared__ float logit[NEXP];
    int t = threadIdx.x;
    int e = t >> 3, c = t & 7;
    float s = 0.f;
    for (int k = c * 128; k < c * 128 + 128; ++k)
        s += g1[k] * gW2[e * 1024 + k];
    part[t] = s;
    __syncthreads();
    if (c == 0) {
        float d = 0.f;
        for (int j = 0; j < 8; ++j) d += part[e * 8 + j];
        logit[e] = d + gb2[e];
    }
    __syncthreads();
    if (t == 0) {
        float mx = logit[0];
        for (int i = 1; i < NEXP; ++i) mx = fmaxf(mx, logit[i]);
        float p[NEXP]; float sum = 0.f;
        for (int i = 0; i < NEXP; ++i) { p[i] = expf(logit[i] - mx); sum += p[i]; }
        for (int i = 0; i < NEXP; ++i) p[i] /= sum;
        int idx[TOPK]; float val[TOPK];
        bool used[NEXP] = {false, false, false, false, false, false, false, false};
        for (int k = 0; k < TOPK; ++k) {
            int best = -1; float bv = -1.0f;
            for (int i = 0; i < NEXP; ++i)
                if (!used[i] && p[i] > bv) { bv = p[i]; best = i; }
            used[best] = true; idx[k] = best; val[k] = bv;
        }
        float m2 = fmaxf(fmaxf(val[0], val[1]), val[2]);
        float g[TOPK]; float ss = 0.f;
        for (int k = 0; k < TOPK; ++k) { g[k] = expf(val[k] - m2); ss += g[k]; }
        for (int k = 0; k < TOPK; ++k) { top_i[k] = idx[k]; gates[k] = g[k] / ss; }
    }
}

// ---------------------------------------------------------------------------
// Finalize: per-expert LayerNorm, gate-weighted sum -> d_out[0:2048].
// ---------------------------------------------------------------------------
__global__ __launch_bounds__(256)
void finalize_kernel(const float* __restrict__ y, const int* __restrict__ top_i,
                     const float* __restrict__ gates,
                     const float* __restrict__ e_gamma, const float* __restrict__ e_beta,
                     float* __restrict__ out)
{
    __shared__ float red[256];
    __shared__ float mu_s[TOPK], rstd_s[TOPK];
    int t = threadIdx.x;
    for (int k = 0; k < TOPK; ++k) {
        float s = 0.f, s2 = 0.f;
        for (int d = t; d < DMODEL; d += 256) {
            float v = y[k * DMODEL + d];
            s += v; s2 += v * v;
        }
        red[t] = s; __syncthreads();
        for (int off = 128; off > 0; off >>= 1) { if (t < off) red[t] += red[t + off]; __syncthreads(); }
        float tot = red[0]; __syncthreads();
        red[t] = s2; __syncthreads();
        for (int off = 128; off > 0; off >>= 1) { if (t < off) red[t] += red[t + off]; __syncthreads(); }
        float tot2 = red[0]; __syncthreads();
        if (t == 0) {
            float mu = tot / (float)DMODEL;
            float var = tot2 / (float)DMODEL - mu * mu;
            mu_s[k] = mu; rstd_s[k] = rsqrtf(var + 1e-5f);
        }
        __syncthreads();
    }
    for (int d = t; d < DMODEL; d += 256) {
        float acc = 0.f;
        for (int k = 0; k < TOPK; ++k) {
            int e = top_i[k];
            float ln = (y[k * DMODEL + d] - mu_s[k]) * rstd_s[k]
                       * e_gamma[e * DMODEL + d] + e_beta[e * DMODEL + d];
            acc += gates[k] * ln;
        }
        out[d] = acc;
    }
}

// ---------------------------------------------------------------------------
// Host launch
// ---------------------------------------------------------------------------
extern "C" void kernel_launch(void* const* d_in, const int* in_sizes, int n_in,
                              void* d_out, int out_size, void* d_ws, size_t ws_size,
                              hipStream_t stream)
{
    const float* feats   = (const float*)d_in[0];
    const float* context = (const float*)d_in[1];
    const float* mod_emb = (const float*)d_in[2];
    const float* rW1     = (const float*)d_in[3];
    const float* rb1     = (const float*)d_in[4];
    const float* rW2     = (const float*)d_in[5];
    const float* rb2     = (const float*)d_in[6];
    const float* rW3     = (const float*)d_in[7];
    const float* rb3     = (const float*)d_in[8];
    const float* W_in    = (const float*)d_in[9];
    const float* b_in    = (const float*)d_in[10];
    const float* W_out   = (const float*)d_in[11];
    const float* b_out   = (const float*)d_in[12];
    const float* gW1     = (const float*)d_in[13];
    const float* gb1     = (const float*)d_in[14];
    const float* gW2     = (const float*)d_in[15];
    const float* gb2     = (const float*)d_in[16];
    const float* eW1     = (const float*)d_in[17];
    const float* eb1     = (const float*)d_in[18];
    const float* eW2     = (const float*)d_in[19];
    const float* eb2     = (const float*)d_in[20];
    const float* e_gamma = (const float*)d_in[21];
    const float* e_beta  = (const float*)d_in[22];

    float* ws  = (float*)d_ws;
    float* out = (float*)d_out;

    float* summ  = ws + OFF_SUMM;
    float* comb  = ws + OFF_COMB;
    float* h1    = ws + OFF_H1;
    float* h2    = ws + OFF_H2;
    float* qkv   = ws + OFF_QKV;
    float* mask  = ws + OFF_MASK;
    float* x     = ws + OFF_X;
    float* apre  = ws + OFF_APRE;
    float* att   = ws + OFF_ATT;
    float* g1    = ws + OFF_G1;
    int*   top_i = (int*)(ws + OFF_TOPI);
    float* gates = ws + OFF_GATES;
    float* hh    = ws + OFF_HH;
    float* yexp  = ws + OFF_Y;

    // 0) zero split-K accumulators (h1|h2|qkv contiguous), embed + summarize
    zero_kernel<<<(ZERO_LEN / 4 + 255) / 256, 256, 0, stream>>>(ws + ZERO_BASE, ZERO_LEN / 4);
    prep_kernel<<<64, 256, 0, stream>>>(feats, context, mod_emb, x, summ, comb);

    // 1) routing MLP layer 1 (WMMA split-K=8) + epilogue relu
    wmma_tokgemm<1><<<dim3(DMODEL / 16, 8), 32, 0, stream>>>(comb, 2 * DMODEL, 512, rW1, h1, DMODEL);
    bias_act_kernel<<<(16 * DMODEL) / 256, 256, 0, stream>>>(h1, rb1, DMODEL, 16 * DMODEL, 1);
    // 2) routing MLP layer 2 (WMMA split-K=8) + epilogue relu
    wmma_tokgemm<1><<<dim3(1024 / 16, 8), 32, 0, stream>>>(h1, DMODEL, 256, rW2, h2, 1024);
    bias_act_kernel<<<(16 * 1024) / 256, 256, 0, stream>>>(h2, rb2, 1024, 16 * 1024, 1);
    // 3) rw + quantile mask (also zeroes attended_pre accumulator)
    routing_finish<<<1, 64, 0, stream>>>(h2, rW3, rb3, mask, out + DMODEL, apre);

    // 4) QKV (WMMA split-K=4, 4 row-tiles share each streamed W_in fragment)
    wmma_tokgemm<4><<<dim3((3 * DMODEL) / 16, 4), 32, 0, stream>>>(x, DMODEL, 512, W_in, qkv, 3 * DMODEL);
    bias_act_kernel<<<(64 * 3 * DMODEL) / 256, 256, 0, stream>>>(qkv, b_in, 3 * DMODEL, 64 * 3 * DMODEL, 0);

    // 5) attention (q at s=0 only), averaged over modalities into apre
    attention_kernel<<<MMOD * NHEAD, 128, 0, stream>>>(qkv, mask, apre);

    // 6) W_out GEMV on averaged token
    gemv_kernel<<<dim3(DMODEL / 8, 1), 256, 0, stream>>>(apre, 0, W_out, 0, b_out, 0,
                                                         nullptr, att, 0, DMODEL, DMODEL, 0);
    // 7) gating
    gemv_kernel<<<dim3(1024 / 8, 1), 256, 0, stream>>>(att, 0, gW1, 0, gb1, 0,
                                                       nullptr, g1, 0, 1024, DMODEL, 1);
    gate_topk<<<1, 64, 0, stream>>>(g1, gW2, gb2, top_i, gates);

    // 8) top-3 experts (indirect expert base from top_i)
    gemv_kernel<<<dim3((2 * DMODEL) / 8, TOPK), 256, 0, stream>>>(
        att, 0, eW1, (long long)(2 * DMODEL) * DMODEL, eb1, 2 * DMODEL,
        top_i, hh, 2 * DMODEL, 2 * DMODEL, DMODEL, 2);
    gemv_kernel<<<dim3(DMODEL / 8, TOPK), 256, 0, stream>>>(
        hh, 2 * DMODEL, eW2, (long long)DMODEL * (2 * DMODEL), eb2, DMODEL,
        top_i, yexp, DMODEL, DMODEL, 2 * DMODEL, 0);

    // 9) per-expert LayerNorm + gated fuse
    finalize_kernel<<<1, 256, 0, stream>>>(yexp, top_i, gates, e_gamma, e_beta, out);
}
